// Block_18665927868802
// MI455X (gfx1250) — matrix operand order repaired and checked by reading, not compiled, not run on hardware
//
#include <hip/hip_runtime.h>

// ============================================================================
// MI455X (gfx1250) transformer block: LN1 -> causal attn -> +res -> LN2 ->
// top-2 MoE (E=2, capacity dispatch) -> +res, plus aux load-balancing loss.
//
// All matmuls run on v_wmma_f32_16x16x32_bf16 (fp32 accum). Weights are
// converted fp32->bf16 once per launch; all activation producers emit bf16
// directly, so GEMM LDS staging is pure data movement. The GEMM is LDS
// double-buffered: one barrier per K-step, global loads for tile k+1 in
// flight while the 8 WMMAs of tile k execute.
// ============================================================================

typedef __attribute__((ext_vector_type(16))) __bf16 v16bf;
typedef __attribute__((ext_vector_type(8)))  float  v8f;

struct Frag256 { unsigned int w[8]; };

static __device__ __forceinline__ unsigned short f2bf(float f) {
  unsigned int u = __float_as_uint(f);
  return (unsigned short)((u + 0x7FFFu + ((u >> 16) & 1u)) >> 16);  // RNE
}
static __device__ __forceinline__ float bf2f(unsigned short h) {
  return __uint_as_float((unsigned int)h << 16);
}

static __device__ __forceinline__ v16bf frag_from(const unsigned short* p0,
                                                  const unsigned short* p1) {
  uint4 a = *(const uint4*)p0;
  uint4 b = *(const uint4*)p1;
  Frag256 t;
  t.w[0] = a.x; t.w[1] = a.y; t.w[2] = a.z; t.w[3] = a.w;
  t.w[4] = b.x; t.w[5] = b.y; t.w[6] = b.z; t.w[7] = b.w;
  return __builtin_bit_cast(v16bf, t);
}

// A fragment of a 16x32 bf16 tile staged as sm[row][k] (row stride ldk):
// lanes 0-15: row = lane,    K = 0..7 and 16..23
// lanes 16-31: row = lane-16, K = 8..15 and 24..31        (ISA 7.12.2)
static __device__ __forceinline__ v16bf load_frag_a(const unsigned short* sm,
                                                    int row0, int ldk, int lane) {
  const unsigned short* base =
      sm + (row0 + (lane & 15)) * ldk + ((lane & 16) ? 8 : 0);
  return frag_from(base, base + 16);
}

// B fragment of a 32x16 bf16 tile staged K-contiguous as sm[col][k]:
// lanes 0-15: col = lane, K = 0..7 and 16..23; lanes 16-31: col = lane-16, K = 8..15 and 24..31
static __device__ __forceinline__ v16bf load_frag_b(const unsigned short* sm,
                                                    int col0, int ldk, int lane) {
  const unsigned short* base =
      sm + (col0 + (lane & 15)) * ldk + ((lane & 16) ? 8 : 0);
  return frag_from(base, base + 16);
}

static __device__ __forceinline__ v8f wmma_bf16(v16bf a, v16bf b, v8f c) {
  return __builtin_amdgcn_wmma_f32_16x16x32_bf16(false, a, false, b, (short)0, c,
                                                 false, false);
}

static __device__ __forceinline__ void scatter8(unsigned short* dst, int stride,
                                                uint4 v) {
  dst[0 * stride] = (unsigned short)(v.x & 0xffffu);
  dst[1 * stride] = (unsigned short)(v.x >> 16);
  dst[2 * stride] = (unsigned short)(v.y & 0xffffu);
  dst[3 * stride] = (unsigned short)(v.y >> 16);
  dst[4 * stride] = (unsigned short)(v.z & 0xffffu);
  dst[5 * stride] = (unsigned short)(v.z >> 16);
  dst[6 * stride] = (unsigned short)(v.w & 0xffffu);
  dst[7 * stride] = (unsigned short)(v.w >> 16);
}

// ---------------------------------------------------------------------------
// fp32 -> bf16 bulk convert (n % 4 == 0 for all call sites)
// ---------------------------------------------------------------------------
__global__ __launch_bounds__(256) void f32_to_bf16_kernel(
    const float* __restrict__ in, unsigned short* __restrict__ out, size_t n) {
  size_t i = ((size_t)blockIdx.x * blockDim.x + threadIdx.x) * 4;
  size_t stride = (size_t)gridDim.x * blockDim.x * 4;
  for (; i < n; i += stride) {
    float4 f = *(const float4*)(in + i);
    uint2 o;
    o.x = (unsigned int)f2bf(f.x) | ((unsigned int)f2bf(f.y) << 16);
    o.y = (unsigned int)f2bf(f.z) | ((unsigned int)f2bf(f.w) << 16);
    *(uint2*)(out + i) = o;
  }
}

__global__ void zero_u32(unsigned int* __restrict__ p, size_t n) {
  size_t i = (size_t)blockIdx.x * blockDim.x + threadIdx.x;
  size_t stride = (size_t)gridDim.x * blockDim.x;
  for (; i < n; i += stride) p[i] = 0u;
}

// ---------------------------------------------------------------------------
// LayerNorm: one 256-thread block per row of C=768; fp32 in, bf16 out.
// ---------------------------------------------------------------------------
__global__ __launch_bounds__(256) void ln_kernel(const float* __restrict__ x,
                                                 const float* __restrict__ w,
                                                 const float* __restrict__ bb,
                                                 unsigned short* __restrict__ out,
                                                 int C) {
  int row = blockIdx.x;
  const float* xr = x + (size_t)row * C;
  unsigned short* orow = out + (size_t)row * C;
  __shared__ float red[8];
  __shared__ float stat[2];
  int tid = threadIdx.x, lane = tid & 31, wv = tid >> 5;

  float s = 0.f;
  for (int c = tid; c < C; c += 256) s += xr[c];
  for (int o = 1; o < 32; o <<= 1) s += __shfl_xor(s, o, 32);
  if (lane == 0) red[wv] = s;
  __syncthreads();
  if (tid == 0) {
    float t = 0.f;
    for (int i = 0; i < 8; i++) t += red[i];
    stat[0] = t / (float)C;
  }
  __syncthreads();
  float mean = stat[0];

  float v = 0.f;
  for (int c = tid; c < C; c += 256) { float d = xr[c] - mean; v += d * d; }
  for (int o = 1; o < 32; o <<= 1) v += __shfl_xor(v, o, 32);
  __syncthreads();
  if (lane == 0) red[wv] = v;
  __syncthreads();
  if (tid == 0) {
    float t = 0.f;
    for (int i = 0; i < 8; i++) t += red[i];
    stat[1] = rsqrtf(t / (float)C + 1e-5f);
  }
  __syncthreads();
  float rstd = stat[1];
  for (int c = tid; c < C; c += 256)
    orow[c] = f2bf((xr[c] - mean) * rstd * w[c] + bb[c]);
}

// ---------------------------------------------------------------------------
// bf16-WMMA GEMM: out[M,N] = act(Abf[M,K] @ Bbf[K,N] + bias) + res
// BM=128, BN=128, BK=32; 8 waves, each a 64x32 patch (4x2 tiles, 8 wmma /
// K-step). LDS double-buffered: one barrier per K-step; tile k+1 global
// loads overlap the WMMAs of tile k; prefetch warms L2 for tile k+2.
// bias/res may be null; out is fp32 or bf16 (out_bf).
// M%128==0, N%128==0, K%32==0 at all call sites.
// ---------------------------------------------------------------------------
#define GBM 128
#define GBN 128
#define GBK 32

__global__ __launch_bounds__(256) void gemm_bf16_kernel(
    const unsigned short* __restrict__ Abf, const unsigned short* __restrict__ Bbf,
    const float* __restrict__ bias, const float* __restrict__ res,
    void* __restrict__ out, int M, int N, int K, int act, int out_bf) {
  __shared__ unsigned short As[2][GBM * GBK] __attribute__((aligned(16)));
  __shared__ unsigned short Bs[2][GBN * GBK] __attribute__((aligned(16)));

  int tid = threadIdx.x, lane = tid & 31, wave = tid >> 5;
  int wm = wave & 1, wn = wave >> 1;  // 2 x 4 wave grid; wave tile 64(m) x 32(n)
  int m0 = blockIdx.y * GBM, n0 = blockIdx.x * GBN;

  // Per-thread staging coordinates: 2 x uint4 (16 bf16) for each of A and B.
  int ra[2], ca[2], kB[2], nB[2];
#pragma unroll
  for (int t = 0; t < 2; t++) {
    int i = tid * 8 + t * 2048;
    ra[t] = i >> 5;  ca[t] = i & 31;    // A tile: 128 rows x 32 k
    kB[t] = i >> 7;  nB[t] = i & 127;   // B tile: 32 k x 128 n
  }

  v8f acc[4][2];
#pragma unroll
  for (int ti = 0; ti < 4; ti++) { acc[ti][0] = {}; acc[ti][1] = {}; }

  uint4 aReg[2], bReg[2];

  // Prologue: stage tile 0 into buffer 0.
#pragma unroll
  for (int t = 0; t < 2; t++) {
    aReg[t] = *(const uint4*)(Abf + (size_t)(m0 + ra[t]) * K + ca[t]);
    bReg[t] = *(const uint4*)(Bbf + (size_t)kB[t] * N + n0 + nB[t]);
  }
#pragma unroll
  for (int t = 0; t < 2; t++) {
    *(uint4*)(&As[0][ra[t] * GBK + ca[t]]) = aReg[t];
    scatter8(&Bs[0][nB[t] * GBK + kB[t]], GBK, bReg[t]);
  }
  __syncthreads();

  int KT = K / GBK;
  for (int kt = 0; kt < KT; kt++) {
    int cur = kt & 1;
    int k1 = (kt + 1) * GBK;
    // Issue next-tile global loads first so they overlap the WMMAs below.
    if (k1 < K) {
#pragma unroll
      for (int t = 0; t < 2; t++) {
        const unsigned short* sa = Abf + (size_t)(m0 + ra[t]) * K + k1 + ca[t];
        const unsigned short* sb = Bbf + (size_t)(k1 + kB[t]) * N + n0 + nB[t];
        aReg[t] = *(const uint4*)sa;
        bReg[t] = *(const uint4*)sb;
        __builtin_prefetch(sa + GBK, 0, 0);             // tile k+2 -> L2
        __builtin_prefetch(sb + (size_t)GBK * N, 0, 0);
      }
    }

    v16bf af[4], bfr[2];
#pragma unroll
    for (int ti = 0; ti < 4; ti++)
      af[ti] = load_frag_a(&As[cur][0], wm * 64 + ti * 16, GBK, lane);
#pragma unroll
    for (int tj = 0; tj < 2; tj++)
      bfr[tj] = load_frag_b(&Bs[cur][0], wn * 32 + tj * 16, GBK, lane);
#pragma unroll
    for (int ti = 0; ti < 4; ti++)
#pragma unroll
      for (int tj = 0; tj < 2; tj++)
        acc[ti][tj] = wmma_bf16(af[ti], bfr[tj], acc[ti][tj]);

    // Store staged registers into the other buffer.
    if (k1 < K) {
      int nxt = cur ^ 1;
#pragma unroll
      for (int t = 0; t < 2; t++) {
        *(uint4*)(&As[nxt][ra[t] * GBK + ca[t]]) = aReg[t];
        scatter8(&Bs[nxt][nB[t] * GBK + kB[t]], GBK, bReg[t]);
      }
    }
    __syncthreads();
  }

  // Epilogue. C layout: lane l holds n = l&15; VGPR i holds m = i + (l>=16?8:0)
  int hsel = (lane >> 4) * 8;
  int nl = lane & 15;
  float* outf = (float*)out;
  unsigned short* outh = (unsigned short*)out;
#pragma unroll
  for (int ti = 0; ti < 4; ti++)
#pragma unroll
    for (int tj = 0; tj < 2; tj++)
#pragma unroll
      for (int i = 0; i < 8; i++) {
        int gm = m0 + wm * 64 + ti * 16 + i + hsel;
        int gn = n0 + wn * 32 + tj * 16 + nl;
        float v = acc[ti][tj][i];
        if (bias) v += bias[gn];
        if (act) v = (v >= 0.f) ? v : 0.01f * v;
        if (res) v += res[(size_t)gm * N + gn];
        if (out_bf) outh[(size_t)gm * N + gn] = f2bf(v);
        else        outf[(size_t)gm * N + gn] = v;
      }
}

// ---------------------------------------------------------------------------
// Causal flash attention. One block per (b*H + h, 64-query block), 4 waves.
// qkv (bf16): [B*T, 3*H*64]; q cols [0,768), k [768,1536), v [1536,2304).
// y (bf16): [B, T, H*64].
// ---------------------------------------------------------------------------
#define HD 64

__global__ __launch_bounds__(128) void attn_kernel(
    const unsigned short* __restrict__ qkv, unsigned short* __restrict__ y,
    int B, int T, int H) {
  int bh = blockIdx.x;
  int b = bh / H, hh = bh % H;
  int qbi = blockIdx.y;
  int q0 = qbi * 64;
  int C3 = 3 * H * HD;
  const unsigned short* base = qkv + (size_t)b * T * C3;

  __shared__ unsigned short Qs[64 * 64] __attribute__((aligned(16)));  // [q][hd]
  __shared__ unsigned short Ks[64 * 64] __attribute__((aligned(16)));  // [key][hd]
  __shared__ unsigned short Vs[64 * 64] __attribute__((aligned(16)));  // [hd][key]
  __shared__ unsigned short Ps[64 * 64] __attribute__((aligned(16)));  // [q][key]

  int tid = threadIdx.x, lane = tid & 31, wave = tid >> 5;
  int hsel = (lane >> 4) * 8;
  int nl = lane & 15;

  // Load Q tile once (raw bf16 copies)
  for (int i = tid * 8; i < 64 * 64; i += 128 * 8) {
    int r = i >> 6, c = i & 63;
    *(uint4*)(Qs + r * 64 + c) =
        *(const uint4*)(base + (size_t)(q0 + r) * C3 + hh * HD + c);
  }

  float m_run[8], l_run[8];
  v8f o[4];
#pragma unroll
  for (int i = 0; i < 8; i++) { m_run[i] = -3.0e38f; l_run[i] = 0.f; }
  o[0] = {}; o[1] = {}; o[2] = {}; o[3] = {};
  const float scale = 0.125f;  // 1/sqrt(64)

  for (int kb = 0; kb <= qbi; kb++) {
    int k0 = kb * 64;
    __syncthreads();  // also covers Qs visibility on first iteration
    for (int i = tid * 8; i < 64 * 64; i += 128 * 8) {
      int r = i >> 6, c = i & 63;
      *(uint4*)(Ks + r * 64 + c) =
          *(const uint4*)(base + (size_t)(k0 + r) * C3 + H * HD + hh * HD + c);
      uint4 fv = *(const uint4*)(base + (size_t)(k0 + r) * C3 + 2 * H * HD + hh * HD + c);
      scatter8(Vs + c * 64 + r, 64, fv);  // transpose into [hd][key]
    }
    __syncthreads();

    // S = Q @ K^T for this wave's 16 query rows x 64 keys (4 tiles, K=64)
    v16bf qa0 = load_frag_a(Qs, wave * 16, 64, lane);
    v16bf qa1 = load_frag_a(Qs + 32, wave * 16, 64, lane);
    v8f s[4];
    s[0] = {}; s[1] = {}; s[2] = {}; s[3] = {};
#pragma unroll
    for (int tj = 0; tj < 4; tj++) {
      v16bf kb0 = load_frag_b(Ks, tj * 16, 64, lane);
      v16bf kb1 = load_frag_b(Ks + 32, tj * 16, 64, lane);
      s[tj] = wmma_bf16(qa0, kb0, s[tj]);
      s[tj] = wmma_bf16(qa1, kb1, s[tj]);
    }

    // Scale + causal mask (diagonal block only)
    if (kb == qbi) {
#pragma unroll
      for (int tj = 0; tj < 4; tj++)
#pragma unroll
        for (int i = 0; i < 8; i++) {
          int qg = wave * 16 + i + hsel;
          int kg = tj * 16 + nl;
          s[tj][i] = (kg <= qg) ? s[tj][i] * scale : -1.0e30f;
        }
    } else {
#pragma unroll
      for (int tj = 0; tj < 4; tj++)
#pragma unroll
        for (int i = 0; i < 8; i++) s[tj][i] *= scale;
    }

    // Row-wise online softmax (each row lives in one 16-lane half)
    float rm[8];
#pragma unroll
    for (int i = 0; i < 8; i++)
      rm[i] = fmaxf(fmaxf(s[0][i], s[1][i]), fmaxf(s[2][i], s[3][i]));
#pragma unroll
    for (int off = 1; off < 16; off <<= 1)
#pragma unroll
      for (int i = 0; i < 8; i++) rm[i] = fmaxf(rm[i], __shfl_xor(rm[i], off, 32));

    float fac[8];
#pragma unroll
    for (int i = 0; i < 8; i++) {
      float mn = fmaxf(m_run[i], rm[i]);
      fac[i] = __expf(m_run[i] - mn);
      m_run[i] = mn;
    }

    float rs[8];
#pragma unroll
    for (int i = 0; i < 8; i++) rs[i] = 0.f;
#pragma unroll
    for (int tj = 0; tj < 4; tj++)
#pragma unroll
      for (int i = 0; i < 8; i++) {
        float p = __expf(s[tj][i] - m_run[i]);
        s[tj][i] = p;
        rs[i] += p;
      }
#pragma unroll
    for (int off = 1; off < 16; off <<= 1)
#pragma unroll
      for (int i = 0; i < 8; i++) rs[i] += __shfl_xor(rs[i], off, 32);
#pragma unroll
    for (int i = 0; i < 8; i++) l_run[i] = l_run[i] * fac[i] + rs[i];

#pragma unroll
    for (int tj = 0; tj < 4; tj++)
#pragma unroll
      for (int i = 0; i < 8; i++) o[tj][i] *= fac[i];

    // Spill P (bf16) into fragment-A layout, then O += P @ V
#pragma unroll
    for (int tj = 0; tj < 4; tj++)
#pragma unroll
      for (int i = 0; i < 8; i++)
        Ps[(wave * 16 + i + hsel) * 64 + tj * 16 + nl] = f2bf(s[tj][i]);
    __syncthreads();

    v16bf pa0 = load_frag_a(Ps, wave * 16, 64, lane);
    v16bf pa1 = load_frag_a(Ps + 32, wave * 16, 64, lane);
#pragma unroll
    for (int tj = 0; tj < 4; tj++) {
      v16bf vb0 = load_frag_b(Vs, tj * 16, 64, lane);
      v16bf vb1 = load_frag_b(Vs + 32, tj * 16, 64, lane);
      o[tj] = wmma_bf16(pa0, vb0, o[tj]);
      o[tj] = wmma_bf16(pa1, vb1, o[tj]);
    }
  }

  // Normalize and write y (bf16) [b, q, h*64 + d]
#pragma unroll
  for (int tj = 0; tj < 4; tj++)
#pragma unroll
    for (int i = 0; i < 8; i++) {
      int qg = q0 + wave * 16 + i + hsel;
      int dg = tj * 16 + nl;
      y[((size_t)b * T + qg) * (H * HD) + hh * HD + dg] = f2bf(o[tj][i] / l_run[i]);
    }
}

// ---------------------------------------------------------------------------
// Gating: one wave per token. g = softmax(h2 @ wg) over E=2 experts.
// ---------------------------------------------------------------------------
__global__ __launch_bounds__(128) void gating_kernel(
    const unsigned short* __restrict__ h2, const float* __restrict__ wg,
    int* __restrict__ top1, float* __restrict__ g1o, float* __restrict__ g2o,
    float* __restrict__ raw, int C) {
  int token = blockIdx.x * 4 + (threadIdx.x >> 5);
  int lane = threadIdx.x & 31;
  const unsigned short* row = h2 + (size_t)token * C;
  float d0 = 0.f, d1 = 0.f;
  for (int c = lane; c < C; c += 32) {
    float v = bf2f(row[c]);
    d0 += v * wg[c * 2 + 0];
    d1 += v * wg[c * 2 + 1];
  }
  for (int off = 1; off < 32; off <<= 1) {
    d0 += __shfl_xor(d0, off, 32);
    d1 += __shfl_xor(d1, off, 32);
  }
  float mx = fmaxf(d0, d1);
  float e0 = __expf(d0 - mx), e1 = __expf(d1 - mx);
  float inv = 1.f / (e0 + e1);
  float r0 = e0 * inv, r1 = e1 * inv;
  int i1 = (r1 > r0) ? 1 : 0;  // jnp.argmax: first index on tie
  float gate1 = (i1 == 0) ? r0 : r1;
  float gate2 = (i1 == 0) ? r1 : r0;
  float denom = gate1 + gate2 + 1e-9f;
  if (lane == 0) {
    top1[token] = i1;
    g1o[token] = gate1 / denom;
    g2o[token] = gate2 / denom;
    raw[(size_t)token * 2 + 0] = r0;
    raw[(size_t)token * 2 + 1] = r1;
  }
}

// ---------------------------------------------------------------------------
// Per-batch sequential capacity scan (exact cumsum semantics) + aux terms.
// ---------------------------------------------------------------------------
__global__ void scan_kernel(const int* __restrict__ top1, int* __restrict__ slot1,
                            int* __restrict__ slot2, const float* __restrict__ raw,
                            float* __restrict__ contrib, int T, int cap) {
  int b = blockIdx.x;
  if (threadIdx.x != 0) return;
  const int* t1 = top1 + (size_t)b * T;
  int* s1 = slot1 + (size_t)b * T;
  int* s2 = slot2 + (size_t)b * T;
  int c1[2] = {0, 0};
  for (int t = 0; t < T; t++) { int e = t1[t]; s1[t] = c1[e]++; }
  int c2[2] = {c1[0], c1[1]};  // pos2 starts at mask1_count
  for (int t = 0; t < T; t++) {
    int e = 1 - t1[t];
    int s = c2[e]++;
    s2[t] = (s < cap) ? s : -1;
  }
  float p0 = 0.f, p1 = 0.f;
  for (int t = 0; t < T; t++) {
    p0 += raw[((size_t)b * T + t) * 2 + 0];
    p1 += raw[((size_t)b * T + t) * 2 + 1];
  }
  float invT = 1.f / (float)T;
  contrib[b] = (c1[0] * invT) * (p0 * invT) + (c1[1] * invT) * (p1 * invT);
}

// Gather h2 (bf16) rows into zero-initialized expert buffers [E][B][cap][C].
__global__ __launch_bounds__(256) void gather_kernel(
    const unsigned short* __restrict__ h2, const int* __restrict__ top1,
    const int* __restrict__ slot1, const int* __restrict__ slot2,
    unsigned short* __restrict__ ein, int B, int T, int C, int cap) {
  int token = blockIdx.x;
  int b = token / T;
  int e1 = top1[token], s1 = slot1[token], s2 = slot2[token];
  const unsigned int* src = (const unsigned int*)(h2 + (size_t)token * C);
  unsigned int* d1 =
      (unsigned int*)(ein + ((((size_t)e1 * B + b) * cap) + s1) * C);
  for (int c = threadIdx.x; c < C / 2; c += 256) d1[c] = src[c];
  if (s2 >= 0) {
    unsigned int* d2 =
        (unsigned int*)(ein + ((((size_t)(1 - e1) * B + b) * cap) + s2) * C);
    for (int c = threadIdx.x; c < C / 2; c += 256) d2[c] = src[c];
  }
}

// xout += gate1 * eo[e1][b][slot1] + gate2 * eo[e2][b][slot2]
__global__ __launch_bounds__(256) void combine_kernel(
    const float* __restrict__ eo, const int* __restrict__ top1,
    const int* __restrict__ slot1, const int* __restrict__ slot2,
    const float* __restrict__ g1, const float* __restrict__ g2,
    float* __restrict__ xout, int B, int T, int C, int cap) {
  int token = blockIdx.x;
  int b = token / T;
  int e1 = top1[token], s1 = slot1[token], s2 = slot2[token];
  float ga = g1[token], gb = g2[token];
  const float* r1 = eo + ((((size_t)e1 * B + b) * cap) + s1) * C;
  const float* r2 =
      (s2 >= 0) ? eo + ((((size_t)(1 - e1) * B + b) * cap) + s2) * C : nullptr;
  float* dst = xout + (size_t)token * C;
  for (int c = threadIdx.x; c < C; c += 256) {
    float v = ga * r1[c];
    if (r2) v += gb * r2[c];
    dst[c] += v;
  }
}

__global__ void finalize_aux(const float* __restrict__ contrib,
                             float* __restrict__ out, int B, float sc) {
  if (blockIdx.x == 0 && threadIdx.x == 0) {
    float s = 0.f;
    for (int i = 0; i < B; i++) s += contrib[i];
    out[0] = s * sc;  // sc = E*E / (B*E)
  }
}

// ===========================================================================
extern "C" void kernel_launch(void* const* d_in, const int* in_sizes, int n_in,
                              void* d_out, int out_size, void* d_ws, size_t ws_size,
                              hipStream_t stream) {
  (void)in_sizes; (void)n_in; (void)out_size; (void)ws_size;
  const int B = 16, T = 1024, C = 768, H = 12, CAP = 1024;
  const size_t BT = (size_t)B * T;  // 16384 tokens (== B*CAP)

  const float* x     = (const float*)d_in[0];
  const float* ln1_w = (const float*)d_in[1];
  const float* ln1_b = (const float*)d_in[2];
  const float* wqkv  = (const float*)d_in[3];
  const float* bqkv  = (const float*)d_in[4];
  const float* wproj = (const float*)d_in[5];
  const float* bproj = (const float*)d_in[6];
  const float* ln2_w = (const float*)d_in[7];
  const float* ln2_b = (const float*)d_in[8];
  const float* wg    = (const float*)d_in[9];
  const float* w1    = (const float*)d_in[10];
  const float* w2    = (const float*)d_in[11];
  const float* w3    = (const float*)d_in[12];

  float* xout = (float*)d_out;  // [B,T,C] then aux scalar

  // ---- workspace layout (~542 MB) ----
  char* cur = (char*)d_ws;
  auto allocF = [&](size_t n) { float* p = (float*)cur; cur += n * 4; return p; };
  auto allocH = [&](size_t n) {
    unsigned short* p = (unsigned short*)cur; cur += n * 2; return p;
  };
  float* eo            = allocF(2 * BT * C);          // expert outputs (fp32)
  unsigned short* hbuf = allocH(BT * C);              // LN out (reused as h2)
  unsigned short* qkvb = allocH(BT * 3 * C);
  unsigned short* ybuf = allocH(BT * C);
  unsigned short* ein  = allocH(2 * BT * C);
  unsigned short* hhb  = allocH(BT * (size_t)(4 * C));
  unsigned short* hh2b = allocH(BT * (size_t)(4 * C));
  unsigned short* wqkvb  = allocH((size_t)C * 3 * C);
  unsigned short* wprojb = allocH((size_t)C * C);
  unsigned short* w1b = allocH((size_t)2 * C * 4 * C);
  unsigned short* w2b = allocH((size_t)2 * 4 * C * 4 * C);
  unsigned short* w3b = allocH((size_t)2 * 4 * C * C);
  float* g1   = allocF(BT);
  float* g2   = allocF(BT);
  float* raw  = allocF(2 * BT);
  float* ctrb = allocF(B);
  int* top1  = (int*)allocF(BT);
  int* slot1 = (int*)allocF(BT);
  int* slot2 = (int*)allocF(BT);

  // 0) one-time weight conversion fp32 -> bf16
  f32_to_bf16_kernel<<<2048, 256, 0, stream>>>(wqkv, wqkvb, (size_t)C * 3 * C);
  f32_to_bf16_kernel<<<1024, 256, 0, stream>>>(wproj, wprojb, (size_t)C * C);
  f32_to_bf16_kernel<<<2048, 256, 0, stream>>>(w1, w1b, (size_t)2 * C * 4 * C);
  f32_to_bf16_kernel<<<4096, 256, 0, stream>>>(w2, w2b, (size_t)2 * 4 * C * 4 * C);
  f32_to_bf16_kernel<<<2048, 256, 0, stream>>>(w3, w3b, (size_t)2 * 4 * C * C);

  // 1) h = LN1(x) -> bf16
  ln_kernel<<<(int)BT, 256, 0, stream>>>(x, ln1_w, ln1_b, hbuf, C);

  // 2) qkv = h @ wqkv + bqkv -> bf16   [16384 x 2304], K=768
  gemm_bf16_kernel<<<dim3(3 * C / GBN, (int)(BT / GBM)), 256, 0, stream>>>(
      hbuf, wqkvb, bqkv, nullptr, qkvb, (int)BT, 3 * C, C, 0, 1);

  // 3) causal flash attention -> y (bf16)
  attn_kernel<<<dim3(B * H, T / 64), 128, 0, stream>>>(qkvb, ybuf, B, T, H);

  // 4) xout = x + y @ wproj + bproj (fp32, residual fused)
  gemm_bf16_kernel<<<dim3(C / GBN, (int)(BT / GBM)), 256, 0, stream>>>(
      ybuf, wprojb, bproj, x, xout, (int)BT, C, C, 0, 0);

  // 5) h2 = LN2(xout) -> bf16 (reuse hbuf)
  ln_kernel<<<(int)BT, 256, 0, stream>>>(xout, ln2_w, ln2_b, hbuf, C);

  // 6) gating per token
  gating_kernel<<<(int)(BT / 4), 128, 0, stream>>>(hbuf, wg, top1, g1, g2, raw, C);

  // 7) capacity scan per batch (deterministic, no atomics)
  scan_kernel<<<B, 32, 0, stream>>>(top1, slot1, slot2, raw, ctrb, T, CAP);

  // 8) dispatch into expert buffers (bf16)
  zero_u32<<<4096, 256, 0, stream>>>((unsigned int*)ein, BT * (size_t)C);
  gather_kernel<<<(int)BT, 256, 0, stream>>>(hbuf, top1, slot1, slot2, ein,
                                             B, T, C, CAP);

  // 9) expert MLPs (LeakyReLU fused; hh/hh2 stay bf16, final out fp32)
  for (int e = 0; e < 2; e++) {
    const unsigned short* w1e = w1b + (size_t)e * C * (4 * C);
    const unsigned short* w2e = w2b + (size_t)e * (4 * C) * (4 * C);
    const unsigned short* w3e = w3b + (size_t)e * (4 * C) * C;
    const unsigned short* eine = ein + (size_t)e * BT * C;
    float* eoe = eo + (size_t)e * BT * C;
    gemm_bf16_kernel<<<dim3(4 * C / GBN, (int)(BT / GBM)), 256, 0, stream>>>(
        eine, w1e, nullptr, nullptr, hhb, (int)BT, 4 * C, C, 1, 1);
    gemm_bf16_kernel<<<dim3(4 * C / GBN, (int)(BT / GBM)), 256, 0, stream>>>(
        hhb, w2e, nullptr, nullptr, hh2b, (int)BT, 4 * C, 4 * C, 1, 1);
    gemm_bf16_kernel<<<dim3(C / GBN, (int)(BT / GBM)), 256, 0, stream>>>(
        hh2b, w3e, nullptr, nullptr, eoe, (int)BT, C, 4 * C, 0, 0);
  }

  // 10) combine gated expert outputs into xout (residual add)
  combine_kernel<<<(int)BT, 256, 0, stream>>>(eo, top1, slot1, slot2, g1, g2,
                                              xout, B, T, C, CAP);

  // 11) aux loss scalar: mean(density*proxy) * E*E == sum(contrib) * E/B
  finalize_aux<<<1, 32, 0, stream>>>(ctrb, xout + BT * C, B, 2.0f / 16.0f);
}
